// FluxGNN_73366631350578
// MI455X (gfx1250) — compile-verified, matrix operand order checked by reading, not run on hardware
//
#include <hip/hip_runtime.h>
#include <hip/hip_bf16.h>

// ---------------------------------------------------------------------------
// FluxGNN for MI455X (gfx1250, wave32, WMMA + async global->LDS).
// Dominant GEMMs ([*,64] @ [64,32]) use v_wmma_f32_16x16x32_f16:
//   K=64 -> two accumulating WMMAs, N=32 -> two 16-col tiles, f32 accum.
// A and B staged in LDS as f32 rows-of-64-along-K (B transposed [N][K]);
// fragments are built with contiguous ds_load_b128 + v_cvt_pk_rtz_f16_f32.
// Edge gather uses global_load_async_to_lds_b128 (ASYNCcnt path).
// ---------------------------------------------------------------------------

typedef __attribute__((ext_vector_type(16))) _Float16 v16h;
typedef __attribute__((ext_vector_type(8)))  float    v8f;

#define DEVINL __device__ __forceinline__

// Build a 16-bit WMMA fragment (16 halves) from an f32 LDS row of 64 K-values.
// ISA 7.12.2 16-bit A/B layout: dwords 0..3 hold K = {0..7}+8*half,
// dwords 4..7 hold K = {16..23}+8*half  (two contiguous 16B runs).
DEVINL v16h frag_from_f32(const float* rowF, int half, int kb32) {
    union { v16h v; unsigned int u[8]; } F;
    const float* p0 = rowF + kb32 + 8 * half;        // K = kb32 + 8h .. +7
    const float* p1 = p0 + 16;                       // K = kb32 + 16 + 8h .. +7
#pragma unroll
    for (int d = 0; d < 4; ++d) {
        auto t0 = __builtin_amdgcn_cvt_pkrtz(p0[2 * d], p0[2 * d + 1]);
        auto t1 = __builtin_amdgcn_cvt_pkrtz(p1[2 * d], p1[2 * d + 1]);
        F.u[d]     = *(unsigned int*)&t0;
        F.u[4 + d] = *(unsigned int*)&t1;
    }
    return F.v;
}

// out[M*32 + N] = relu( A(16x64) @ B(64x32) + bias[N] )
// ldsA: f32 [16][64] row-major (K along row).  ldsBt: f32 [32][64] (B transposed).
DEVINL void mma_tile_64x32_relu(const float* ldsA, const float* ldsBt,
                                const float* __restrict__ bias,
                                float* out, int lane) {
    int M = lane & 15, half = lane >> 4;
    const float* arow = ldsA + M * 64;
#pragma unroll
    for (int nb = 0; nb < 2; ++nb) {
        int Ncol = (lane & 15) + 16 * nb;
        const float* brow = ldsBt + Ncol * 64;
        v8f c = {};
#pragma unroll
        for (int kb = 0; kb < 2; ++kb) {
            v16h a = frag_from_f32(arow, half, 32 * kb);
            v16h b = frag_from_f32(brow, half, 32 * kb);
            c = __builtin_amdgcn_wmma_f32_16x16x32_f16(
                    false, a, false, b, (short)0, c, false, false);
        }
        float bv = bias[Ncol];
#pragma unroll
        for (int r = 0; r < 8; ++r) {
            int Mo = r + (half ? 8 : 0);             // C layout: ISA 7.12.2
            float v = c[r] + bv;
            out[Mo * 32 + Ncol] = v > 0.f ? v : 0.f;
        }
    }
}

// --------------------------- small helper kernels ---------------------------

__global__ void k_zero(float* __restrict__ p, long long n) {
    long long i = (long long)blockIdx.x * blockDim.x + threadIdx.x;
    long long stride = (long long)gridDim.x * blockDim.x;
    for (; i < n; i += stride) p[i] = 0.f;
}

// h0 = relu(X[N,2] @ W_in[2,32] + b_in)
__global__ void k_input(const float* __restrict__ X, const float* __restrict__ W,
                        const float* __restrict__ b, float* __restrict__ h0, int Nn) {
    int i = blockIdx.x * blockDim.x + threadIdx.x;
    if (i >= Nn * 32) return;
    int n = i >> 5, j = i & 31;
    float v = fmaf(X[n * 2 + 0], W[j], fmaf(X[n * 2 + 1], W[32 + j], b[j]));
    h0[i] = v > 0.f ? v : 0.f;
}

__global__ void k_degree(const int* __restrict__ row, float* __restrict__ deg, int Ee) {
    int e = blockIdx.x * blockDim.x + threadIdx.x;
    if (e < Ee) atomicAdd(&deg[row[e]], 1.0f);
}

// agg[row[e], :] += h[col[e], :]
__global__ void k_scatter(const int* __restrict__ row, const int* __restrict__ col,
                          const float* __restrict__ h, float* __restrict__ agg, int Ee) {
    int i = blockIdx.x * blockDim.x + threadIdx.x;
    if (i >= Ee * 32) return;
    int e = i >> 5, j = i & 31;
    atomicAdd(&agg[(long long)row[e] * 32 + j], h[(long long)col[e] * 32 + j]);
}

// ------------------------- WMMA node-update kernel --------------------------
// h_out = relu( [h | agg/deg] @ W[64,32] + bias )   one 16-node tile per wave
__global__ __launch_bounds__(256)
void k_node_update(const float* __restrict__ h, const float* __restrict__ agg,
                   const float* __restrict__ deg, const float* __restrict__ W,
                   const float* __restrict__ bias, float* __restrict__ hout, int Nn) {
    __shared__ __align__(16) float sBt[32 * 64];     // W transposed [N][K]
    __shared__ __align__(16) float sF[8][16 * 64];   // A tiles, f32

    int lane = threadIdx.x & 31, wv = threadIdx.x >> 5;
    int ntiles = (Nn + 15) >> 4;
    int tile = blockIdx.x * 8 + wv;
    if (tile >= ntiles) tile = ntiles - 1;   // duplicate tail tile: identical work

    // cooperative transposed weight load: W[k][n] -> sBt[n][k]
    for (int i = threadIdx.x; i < 2048; i += 256)
        sBt[(i & 31) * 64 + (i >> 5)] = W[i];

    int m = lane & 15, which = lane >> 4;
    int node = tile * 16 + m;
    if (node >= Nn) node = Nn - 1;
    const float4* src = (const float4*)((which ? agg : h) + (long long)node * 32);
    float scale = which ? 1.0f / fmaxf(deg[node], 1.0f) : 1.0f;
    float4* dst = (float4*)&sF[wv][m * 64 + which * 32];
#pragma unroll
    for (int q = 0; q < 8; ++q) {
        float4 v = src[q];
        v.x *= scale; v.y *= scale; v.z *= scale; v.w *= scale;
        dst[q] = v;
    }
    __syncthreads();
    mma_tile_64x32_relu(sF[wv], sBt, bias, hout + (long long)tile * 512, lane);
}

// --------------------------- WMMA edge-MLP kernel ---------------------------
// hidden = relu([h[row]|h[col]] @ We1 + be1); flux = hidden @ We2 + be2
__global__ __launch_bounds__(256)
void k_edge(const float* __restrict__ h, const int* __restrict__ row,
            const int* __restrict__ col, const float* __restrict__ We1,
            const float* __restrict__ be1, const float* __restrict__ We2,
            const float* __restrict__ be2, float* __restrict__ flux, int Ee) {
    __shared__ __align__(16) float sBt[32 * 64];     // We1 transposed [N][K]
    __shared__ __align__(16) float sF[8][16 * 64];   // edge_feat tiles, f32
    __shared__ __align__(16) float sH[8][16 * 32];   // hidden tiles

    int lane = threadIdx.x & 31, wv = threadIdx.x >> 5;
    int ntiles = (Ee + 15) >> 4;
    int tile = blockIdx.x * 8 + wv;
    if (tile >= ntiles) tile = ntiles - 1;

    for (int i = threadIdx.x; i < 2048; i += 256)
        sBt[(i & 31) * 64 + (i >> 5)] = We1[i];

    // gather: lane = (edge-in-tile m, endpoint which); each lane copies one
    // 128B h-row straight into LDS with 8 async b128 transfers (ASYNCcnt).
    int m = lane & 15, which = lane >> 4;
    long long e = (long long)tile * 16 + m;
    if (e >= Ee) e = Ee - 1;
    int node = which ? col[e] : row[e];
    const char* gsrc = (const char*)(h + (long long)node * 32);
    unsigned lbase = (unsigned)(uintptr_t)&sF[wv][m * 64 + which * 32];
#pragma unroll
    for (int q = 0; q < 8; ++q) {
        unsigned loff = lbase + q * 16;
        unsigned long long ga = (unsigned long long)(uintptr_t)(gsrc + q * 16);
        asm volatile("global_load_async_to_lds_b128 %0, %1, off"
                     :: "v"(loff), "v"(ga) : "memory");
    }
    asm volatile("s_wait_asynccnt 0x0" ::: "memory");
    __syncthreads();

    mma_tile_64x32_relu(sF[wv], sBt, be1, sH[wv], lane);
    __syncthreads();

    if (lane < 16) {
        long long eo = (long long)tile * 16 + lane;
        if (eo < Ee) {
            float s = be2[0];
            const float4* hv = (const float4*)&sH[wv][lane * 32];
            const float4* w4 = (const float4*)We2;
#pragma unroll
            for (int q = 0; q < 8; ++q) {
                float4 a4 = hv[q], b4 = w4[q];
                s = fmaf(a4.x, b4.x, fmaf(a4.y, b4.y,
                        fmaf(a4.z, b4.z, fmaf(a4.w, b4.w, s))));
            }
            flux[eo] = s;
        }
    }
}

// --------------------------------- driver -----------------------------------

extern "C" void kernel_launch(void* const* d_in, const int* in_sizes, int n_in,
                              void* d_out, int out_size, void* d_ws, size_t ws_size,
                              hipStream_t stream) {
    const int Nn = in_sizes[0] / 2;   // F = 2
    const int Ee = in_sizes[1] / 2;

    const float* X     = (const float*)d_in[0];
    const int*   row   = (const int*)d_in[1];
    const int*   col   = row + Ee;
    const float* W_in  = (const float*)d_in[2];
    const float* b_in  = (const float*)d_in[3];
    const float* W_upd = (const float*)d_in[4];   // [2][64][32]
    const float* b_upd = (const float*)d_in[5];   // [2][32]
    const float* We1   = (const float*)d_in[6];   // [64][32]
    const float* be1   = (const float*)d_in[7];
    const float* We2   = (const float*)d_in[8];   // [32]
    const float* be2   = (const float*)d_in[9];
    float* flux = (float*)d_out;

    // workspace layout (all f32)
    float* h_a = (float*)d_ws;                      // N*32
    float* h_b = h_a + (size_t)Nn * 32;             // N*32
    float* agg = h_b + (size_t)Nn * 32;             // N*32
    float* deg = agg + (size_t)Nn * 32;             // N

    // h0 = relu(X @ W_in + b_in)
    k_input<<<(Nn * 32 + 255) / 256, 256, 0, stream>>>(X, W_in, b_in, h_a, Nn);

    // degree
    k_zero<<<1024, 256, 0, stream>>>(deg, (long long)Nn);
    k_degree<<<(Ee + 255) / 256, 256, 0, stream>>>(row, deg, Ee);

    // two message-passing layers (ping-pong h buffers)
    float* hin = h_a; float* hout = h_b;
    int ntilesN = (Nn + 15) >> 4;
    for (int l = 0; l < 2; ++l) {
        k_zero<<<4096, 256, 0, stream>>>(agg, (long long)Nn * 32);
        k_scatter<<<(Ee * 32 + 255) / 256, 256, 0, stream>>>(row, col, hin, agg, Ee);
        k_node_update<<<(ntilesN + 7) / 8, 256, 0, stream>>>(
            hin, agg, deg, W_upd + (size_t)l * 2048, b_upd + l * 32, hout, Nn);
        float* t = hin; hin = hout; hout = t;
    }

    // edge MLP -> flux
    int ntilesE = (Ee + 15) >> 4;
    k_edge<<<(ntilesE + 7) / 8, 256, 0, stream>>>(
        hin, row, col, We1, be1, We2, be2, flux, Ee);
}
